// SelfAttention_78915729097413
// MI455X (gfx1250) — matrix operand order repaired and checked by reading, not compiled, not run on hardware
//
#include <hip/hip_runtime.h>

#define B_    4
#define C_    256
#define KD    32      // low-res channels (c/8)
#define N_    4096    // H*W
#define OROWS 320     // KD + KD + C_

// padded LDS strides (elements) - keep 16B alignment (32B for SGT_S/SPT_S), spread banks
#define SGT_S 48
#define SPT_S 80
#define SHV_S 72

typedef __attribute__((ext_vector_type(16))) __bf16 v16bf;
typedef __attribute__((ext_vector_type(8)))  __bf16 v8bf;
typedef __attribute__((ext_vector_type(8)))  float  v8f;
typedef int v4i __attribute__((vector_size(16)));   // matches builtin V4i

__device__ __forceinline__ v8f wmma_bf16(v16bf a, v16bf b, v8f c) {
  return __builtin_amdgcn_wmma_f32_16x16x32_bf16(
      false, a, false, b, (short)0, c, false, false);
}

__device__ __forceinline__ v16bf concat8(v8bf lo, v8bf hi) {
  return __builtin_shufflevector(lo, hi, 0, 1, 2, 3, 4, 5, 6, 7,
                                 8, 9, 10, 11, 12, 13, 14, 15);
}

// order-preserving float <-> uint key (for atomicMax on LDS)
__device__ __forceinline__ unsigned f2key(float f) {
  unsigned b = __float_as_uint(f);
  return (b & 0x80000000u) ? ~b : (b | 0x80000000u);
}
__device__ __forceinline__ float key2f(unsigned k) {
  unsigned b = (k & 0x80000000u) ? (k & 0x7FFFFFFFu) : ~k;
  return __uint_as_float(b);
}

// ---- CDNA5 async global->LDS staging (ASYNCcnt-tracked), with fallback ----
#if __has_builtin(__builtin_amdgcn_global_load_async_to_lds_b128)
#define HAVE_ASYNC_LDS 1
#else
#define HAVE_ASYNC_LDS 0
#endif

__device__ __forceinline__ void stage16(const __bf16* g, __bf16* l) {
#if HAVE_ASYNC_LDS
  __builtin_amdgcn_global_load_async_to_lds_b128(
      (__attribute__((address_space(1))) v4i*)g,
      (__attribute__((address_space(3))) v4i*)l, 0, 0);
#else
  *(v8bf*)l = *(const v8bf*)g;
#endif
}

__device__ __forceinline__ void wait_async_all() {
#if HAVE_ASYNC_LDS && __has_builtin(__builtin_amdgcn_s_wait_asynccnt)
  __builtin_amdgcn_s_wait_asynccnt(0);
#endif
}
__device__ __forceinline__ void wait_async_keep8() {
#if HAVE_ASYNC_LDS && __has_builtin(__builtin_amdgcn_s_wait_asynccnt)
  __builtin_amdgcn_s_wait_asynccnt(8);
#endif
}

// stage h(:, i0..i0+63): 256 rows x 128B -> LDS, 8 x 16B chunks per thread
__device__ __forceinline__ void stage_hv(const __bf16* hv, __bf16* dst,
                                         int b, int i0, int tid) {
  const int sub = tid & 7;
  const int r0  = tid >> 3;
#pragma unroll
  for (int k = 0; k < 8; ++k) {
    const int row = r0 + k * 32;
    stage16(hv + ((size_t)b * C_ + row) * N_ + i0 + sub * 8,
            dst + row * SHV_S + sub * 8);
  }
}

// ---------------------------------------------------------------------------
// Kernel 1: fused projections. rows 0..31 = f, 32..63 = g, 64..319 = h.
//   fvT : (B, N, KD) bf16     gvT : (B, N, KD) bf16     hv : (B, C, N) bf16
// ---------------------------------------------------------------------------
__global__ __launch_bounds__(256) void proj_kernel(
    const float* __restrict__ x,
    const float* __restrict__ f_w, const float* __restrict__ f_b,
    const float* __restrict__ g_w, const float* __restrict__ g_b,
    const float* __restrict__ h_w, const float* __restrict__ h_b,
    __bf16* __restrict__ fvT, __bf16* __restrict__ gvT, __bf16* __restrict__ hv)
{
  const int lane = threadIdx.x & 31;
  const int wave = threadIdx.x >> 5;
  const int tid  = blockIdx.x * 8 + wave;
  const int jt   = tid & 255;
  const int ot   = (tid >> 8) % (OROWS / 16);
  const int b    = tid / (256 * (OROWS / 16));
  const int n0   = jt * 16;
  const int l15  = lane & 15;
  const int hi   = lane >> 4;

  // o0 is wave-uniform; a 16-row tile never straddles the f/g/h boundary.
  const int o0  = __builtin_amdgcn_readfirstlane(ot * 16);
  const int seg = (o0 >= 2 * KD) ? 2 : (o0 >= KD) ? 1 : 0;

  const int mrow = o0 + l15;
  const float* wrow = (seg == 0) ? (f_w + mrow * C_)
                    : (seg == 1) ? (g_w + (mrow - KD) * C_)
                                 : (h_w + (mrow - 2 * KD) * C_);
  const float4* w4 = (const float4*)wrow;

  const float* xb = x + (size_t)b * C_ * N_;
  v8f acc = {};

#pragma unroll 1
  for (int kk = 0; kk < C_; kk += 32) {
    // A-frag: 4 x float4 vector loads (rows contiguous), convert to bf16
    const int q = (kk >> 2) + hi * 2;
    const float4 w0 = w4[q], w1 = w4[q + 1], w2 = w4[q + 4], w3 = w4[q + 5];
    v16bf a;
    a[0]  = (__bf16)w0.x; a[1]  = (__bf16)w0.y; a[2]  = (__bf16)w0.z; a[3]  = (__bf16)w0.w;
    a[4]  = (__bf16)w1.x; a[5]  = (__bf16)w1.y; a[6]  = (__bf16)w1.z; a[7]  = (__bf16)w1.w;
    a[8]  = (__bf16)w2.x; a[9]  = (__bf16)w2.y; a[10] = (__bf16)w2.z; a[11] = (__bf16)w2.w;
    a[12] = (__bf16)w3.x; a[13] = (__bf16)w3.y; a[14] = (__bf16)w3.z; a[15] = (__bf16)w3.w;

    v16bf bb;   // B layout: lane = column, K striped (strided column reads)
#pragma unroll
    for (int e = 0; e < 16; ++e) {
      const int kc = kk + (e & 7) + hi * 8 + (e >> 3) * 16;
      bb[e] = (__bf16)xb[(size_t)kc * N_ + (n0 + l15)];
    }
    acc = wmma_bf16(a, bb, acc);
  }

  // ---- store epilogue: wave-uniform segment, vectorized bias + stores ----
  const int n = n0 + l15;
  if (seg == 0) {
    // rows o0+hi*8 .. +7 are contiguous in fvT's (n,k)-major layout
    const float4* b4 = (const float4*)(f_b + o0 + hi * 8);
    const float4 b0 = b4[0], b1 = b4[1];
    v8bf st;
    st[0] = (__bf16)(acc[0] + b0.x); st[1] = (__bf16)(acc[1] + b0.y);
    st[2] = (__bf16)(acc[2] + b0.z); st[3] = (__bf16)(acc[3] + b0.w);
    st[4] = (__bf16)(acc[4] + b1.x); st[5] = (__bf16)(acc[5] + b1.y);
    st[6] = (__bf16)(acc[6] + b1.z); st[7] = (__bf16)(acc[7] + b1.w);
    *(v8bf*)(fvT + ((size_t)b * N_ + n) * KD + o0 + hi * 8) = st;
  } else if (seg == 1) {
    const int og = o0 - KD;
    const float4* b4 = (const float4*)(g_b + og + hi * 8);
    const float4 b0 = b4[0], b1 = b4[1];
    v8bf st;
    st[0] = (__bf16)(acc[0] + b0.x); st[1] = (__bf16)(acc[1] + b0.y);
    st[2] = (__bf16)(acc[2] + b0.z); st[3] = (__bf16)(acc[3] + b0.w);
    st[4] = (__bf16)(acc[4] + b1.x); st[5] = (__bf16)(acc[5] + b1.y);
    st[6] = (__bf16)(acc[6] + b1.z); st[7] = (__bf16)(acc[7] + b1.w);
    *(v8bf*)(gvT + ((size_t)b * N_ + n) * KD + og + hi * 8) = st;
  } else {
    const int oh = o0 - 2 * KD;
    const float4* b4 = (const float4*)(h_b + oh + hi * 8);
    const float4 b0 = b4[0], b1 = b4[1];
    const float bias[8] = {b0.x, b0.y, b0.z, b0.w, b1.x, b1.y, b1.z, b1.w};
    __bf16* hrow = hv + ((size_t)b * C_ + (oh + hi * 8)) * N_ + n;
#pragma unroll
    for (int r = 0; r < 8; ++r)
      hrow[(size_t)r * N_] = (__bf16)(acc[r] + bias[r]);
  }
}

// ---------------------------------------------------------------------------
// Kernel 2: flash-style attention + epilogue.
// One workgroup (8 waves) per (batch, 64-column block).
// ---------------------------------------------------------------------------
__global__ __launch_bounds__(256) void attn_kernel(
    const __bf16* __restrict__ fvT,
    const __bf16* __restrict__ gvT,
    const __bf16* __restrict__ hv,
    const float* __restrict__ t_in,
    const float* __restrict__ gamma_p,
    float* __restrict__ out)
{
  __shared__ __align__(32) __bf16 sGT[64 * SGT_S];       // gT block (col-major)
  __shared__ __align__(32) __bf16 sPT[64 * SPT_S];       // P block (col-major)
  __shared__ __align__(32) __bf16 sHV[2][C_ * SHV_S];    // double-buffered h tile
  __shared__ unsigned sMu[64];
  __shared__ float    sM[64];
  __shared__ float    sL[64];

  const int tid  = threadIdx.x;
  const int lane = tid & 31;
  const int wave = tid >> 5;
  const int l15  = lane & 15;
  const int hi   = lane >> 4;
  const int b    = blockIdx.x >> 6;
  const int j0   = (blockIdx.x & 63) * 64;

  // stage gT block (64 cols x 32 k; contiguous 4KB in gvT) -> async to LDS
  {
    const int col = tid >> 2, part = tid & 3;
    stage16(gvT + ((size_t)b * N_ + (j0 + col)) * KD + part * 8,
            &sGT[col * SGT_S + part * 8]);
  }
  if (tid < 64) { sMu[tid] = 0u; sL[tid] = 0.0f; }
  wait_async_all();
  __syncthreads();

  // ---------------- pass 1: column maxima over i ----------------
#pragma unroll 1
  for (int i0 = 0; i0 < N_; i0 += 64) {
#pragma unroll
    for (int t = wave * 2; t < wave * 2 + 2; ++t) {
      const int it = t >> 2, jtt = (t & 3) * 16;
      const v8bf* fr =
          (const v8bf*)(fvT + ((size_t)b * N_ + (i0 + it * 16 + l15)) * KD);
      const v16bf a  = concat8(fr[2 * hi], fr[2 * hi + 1]);
      const v16bf bb = *(const v16bf*)&sGT[(jtt + l15) * SGT_S + hi * 16];
      v8f s = {};
      s = wmma_bf16(a, bb, s);
      float mx = s[0];
#pragma unroll
      for (int r = 1; r < 8; ++r) mx = fmaxf(mx, s[r]);
      atomicMax(&sMu[jtt + l15], f2key(mx));
    }
  }
  __syncthreads();
  if (tid < 64) sM[tid] = key2f(sMu[tid]);
  stage_hv(hv, &sHV[0][0], b, 0, tid);   // kick off first h tile
  __syncthreads();

  // ---------------- pass 2: P and acc += h * P ----------------
  v8f acc[2][4];
  {
    v8f z = {};
#pragma unroll
    for (int ct = 0; ct < 2; ++ct)
#pragma unroll
      for (int j = 0; j < 4; ++j) acc[ct][j] = z;
  }

#pragma unroll 1
  for (int i0 = 0; i0 < N_; i0 += 64) {
    const int cb = (i0 >> 6) & 1;
    __bf16* bufc = &sHV[cb][0];
    __bf16* bufn = &sHV[cb ^ 1][0];

    // S -> P for this wave's two tiles (store transposed, packed 16B)
#pragma unroll
    for (int t = wave * 2; t < wave * 2 + 2; ++t) {
      const int it = t >> 2, jtt = (t & 3) * 16;
      const v8bf* fr =
          (const v8bf*)(fvT + ((size_t)b * N_ + (i0 + it * 16 + l15)) * KD);
      const v16bf a  = concat8(fr[2 * hi], fr[2 * hi + 1]);
      const v16bf bb = *(const v16bf*)&sGT[(jtt + l15) * SGT_S + hi * 16];
      v8f s = {};
      s = wmma_bf16(a, bb, s);
      const float m = sM[jtt + l15];
      float lsum = 0.0f;
      v8bf pv;
#pragma unroll
      for (int r = 0; r < 8; ++r) {
        const float p = __expf(s[r] - m);
        lsum += p;
        pv[r] = (__bf16)p;
      }
      *(v8bf*)&sPT[(jtt + l15) * SPT_S + it * 16 + hi * 8] = pv;
      atomicAdd(&sL[jtt + l15], lsum);
    }

    const bool have_next = (i0 + 64 < N_);
    if (have_next) {
      stage_hv(hv, bufn, b, i0 + 64, tid);   // prefetch next tile (async)
      __builtin_prefetch(fvT + ((size_t)b * N_ + (i0 + 64)) * KD + lane * 8, 0, 0);
      wait_async_keep8();                    // current tile complete (in-order)
    } else {
      wait_async_all();
    }
    __syncthreads();

    // acc(256x64) += h(:, i-block) * P : 2 K-steps of 32, 8 WMMAs each
#pragma unroll
    for (int kk2 = 0; kk2 < 64; kk2 += 32) {
      v16bf aA[2], bB[4];
#pragma unroll
      for (int ct = 0; ct < 2; ++ct) {
        const int c0 = wave * 32 + ct * 16;
        const v8bf* hr = (const v8bf*)&bufc[(c0 + l15) * SHV_S + kk2];
        aA[ct] = concat8(hr[2 * hi], hr[2 * hi + 1]);
      }
#pragma unroll
      for (int j = 0; j < 4; ++j)
        bB[j] = *(const v16bf*)&sPT[(j * 16 + l15) * SPT_S + kk2 + hi * 16];
#pragma unroll
      for (int ct = 0; ct < 2; ++ct)
#pragma unroll
        for (int j = 0; j < 4; ++j)
          acc[ct][j] = wmma_bf16(aA[ct], bB[j], acc[ct][j]);
    }
    __syncthreads();
  }

  // ---------------- epilogue: out = gamma * acc / l + t_in ----------------
  const float gamma = gamma_p[0];
#pragma unroll
  for (int ct = 0; ct < 2; ++ct) {
    const int c0 = wave * 32 + ct * 16;
#pragma unroll
    for (int j = 0; j < 4; ++j) {
      const int   jcol = j0 + j * 16 + l15;
      const float invl = 1.0f / sL[j * 16 + l15];
#pragma unroll
      for (int r = 0; r < 8; ++r) {
        const int    c   = c0 + r + hi * 8;
        const size_t idx = ((size_t)b * C_ + c) * N_ + jcol;
        out[idx] = gamma * acc[ct][j][r] * invl + t_in[idx];
      }
    }
  }
}

// ---------------------------------------------------------------------------
extern "C" void kernel_launch(void* const* d_in, const int* in_sizes, int n_in,
                              void* d_out, int out_size, void* d_ws, size_t ws_size,
                              hipStream_t stream) {
  (void)in_sizes; (void)n_in; (void)out_size; (void)ws_size;
  const float* t_in  = (const float*)d_in[0];
  const float* f_w   = (const float*)d_in[1];
  const float* f_b   = (const float*)d_in[2];
  const float* g_w   = (const float*)d_in[3];
  const float* g_b   = (const float*)d_in[4];
  const float* h_w   = (const float*)d_in[5];
  const float* h_b   = (const float*)d_in[6];
  const float* gamma = (const float*)d_in[7];
  float*       out   = (float*)d_out;

  __bf16* fvT = (__bf16*)d_ws;                   // B*N*KD  (1 MB)
  __bf16* gvT = fvT + (size_t)B_ * N_ * KD;      // B*N*KD  (1 MB)
  __bf16* hvp = gvT + (size_t)B_ * N_ * KD;      // B*C*N   (8 MB)

  proj_kernel<<<20480 / 8, 256, 0, stream>>>(t_in, f_w, f_b, g_w, g_b,
                                             h_w, h_b, fvT, gvT, hvp);
  attn_kernel<<<B_ * (N_ / 64), 256, 0, stream>>>(fvT, gvT, hvp, t_in, gamma, out);
}
